// MixtralDecoderLayer_21182778704251
// MI455X (gfx1250) — matrix operand order repaired and checked by reading, not compiled, not run on hardware
//
#include <hip/hip_runtime.h>
#include <hip/hip_bf16.h>

#define HIDN  1024
#define NHEAD 16
#define NKVH  4
#define HDIM  64
#define NEXP  8
#define INTERN 3584
#define SEQ   2048

typedef __bf16 bf16_t;
typedef __attribute__((ext_vector_type(16))) __bf16 v16bf;
typedef __attribute__((ext_vector_type(8)))  __bf16 v8bf;
typedef __attribute__((ext_vector_type(8)))  float  v8f;
typedef __attribute__((ext_vector_type(4)))  unsigned int v4u_t;
typedef __attribute__((ext_vector_type(8)))  int v8i_t;
typedef __attribute__((ext_vector_type(4)))  int v4i_t;

#define WMMA_BF16(A,B,C) \
  __builtin_amdgcn_wmma_f32_16x16x32_bf16(false,(A),false,(B),(short)0,(C),false,false)

// ---------------- fragment loaders (contiguous-K layouts, same for A and B, wave32) ----------------
// A (16x32 bf16, MxK): lane L -> row = L&15; elems 0..15 -> K = 16h+0..15
// (h = L>>4)
__device__ __forceinline__ v16bf load_a_frag(const bf16_t* __restrict__ A, int lda,
                                             int m0, int k0, int lane) {
  const int row  = m0 + (lane & 15);
  const int half = lane >> 4;
  const bf16_t* p = A + (size_t)row * lda + k0 + 16 * half;
  v8bf lo = *(const v8bf*)p;
  v8bf hi = *(const v8bf*)(p + 8);
  v16bf f;
#pragma unroll
  for (int i = 0; i < 8; ++i) { f[i] = lo[i]; f[8 + i] = hi[i]; }
  return f;
}
// B (32x16 bf16, KxN) from row-major W[N,K] in global: lane -> col n0+(L&15),
// 16 contiguous K at k0+16h
__device__ __forceinline__ v16bf load_b_frag(const bf16_t* __restrict__ W, int ldw,
                                             int n0, int k0, int lane) {
  const int col  = n0 + (lane & 15);
  const int half = lane >> 4;
  return *(const v16bf*)(W + (size_t)col * ldw + k0 + 16 * half);
}
// B fragment from an LDS-staged tile laid out [rows(N-local)][64 K-local] bf16
__device__ __forceinline__ v16bf load_b_lds(const bf16_t* Bs, int nLocal, int kLocal,
                                            int lane) {
  const int col  = nLocal + (lane & 15);
  const int half = lane >> 4;
  return *(const v16bf*)(Bs + col * 64 + kLocal + 16 * half);
}

// ---------------- Tensor Data Mover: 2D tile (rows x kElems bf16) -> LDS ----------------
// D# per CDNA5 ISA 8.3/8.4: group0 {count=1, lds_addr, global_addr, type=2},
// group1 {data_size=2B, tensor_dim0=ldK, tensor_dim1=rowsTotal, tile_dim0=kElems,
//         tile_dim1=rows, tensor_dim0_stride=ldK}
__device__ __forceinline__ void tdm_load_2d(unsigned lds_off, const void* gptr,
                                            unsigned rows, unsigned kElems,
                                            unsigned ldK, unsigned rowsTotal) {
  const unsigned long long ga = (unsigned long long)(size_t)gptr;
  v4u_t g0;
  g0[0] = 1u;                                               // count=1
  g0[1] = lds_off;                                          // lds_addr (bytes)
  g0[2] = (unsigned)ga;                                     // global_addr[31:0]
  g0[3] = (unsigned)((ga >> 32) & 0x01FFFFFFu) | (2u << 30);// addr[56:32] | type=2
  v8i_t g1;
  g1[0] = (int)(1u << 16);                                  // data_size=1 -> 2 bytes
  g1[1] = (int)((ldK & 0xFFFFu) << 16);                     // tensor_dim0[15:0]
  g1[2] = (int)(((ldK >> 16) & 0xFFFFu) | ((rowsTotal & 0xFFFFu) << 16));
  g1[3] = (int)(((rowsTotal >> 16) & 0xFFFFu) | ((kElems & 0xFFFFu) << 16)); // tile_dim0
  g1[4] = (int)(rows & 0xFFFFu);                            // tile_dim1 (tile_dim2=0)
  g1[5] = (int)ldK;                                         // tensor_dim0_stride lo32
  g1[6] = 0;                                                // stride hi16 | dim1_stride lo16
  g1[7] = 0;
  v4i_t z4 = {};
#if __clang_major__ >= 23
  v8i_t z8 = {};
  __builtin_amdgcn_tensor_load_to_lds(g0, g1, z4, z4, z8, 0);
#else
  __builtin_amdgcn_tensor_load_to_lds(g0, g1, z4, z4, 0);
#endif
}

// ---------------- elementwise f32 -> bf16 ----------------
__global__ void f32_to_bf16_kernel(const float* __restrict__ in,
                                   bf16_t* __restrict__ out, size_t n) {
  size_t i = (size_t)blockIdx.x * blockDim.x + threadIdx.x;
  size_t stride = (size_t)gridDim.x * blockDim.x;
  for (; i < n; i += stride) out[i] = (bf16_t)in[i];
}

// ---------------- RMSNorm (f32 in, bf16 out) ----------------
__global__ __launch_bounds__(256)
void rmsnorm_kernel(const float* __restrict__ X, const float* __restrict__ W,
                    bf16_t* __restrict__ O) {
  const int t = blockIdx.x;
  const float* x = X + (size_t)t * HIDN;
  float ss = 0.f;
  for (int i = threadIdx.x; i < HIDN; i += 256) { float v = x[i]; ss += v * v; }
#pragma unroll
  for (int off = 16; off; off >>= 1) ss += __shfl_xor(ss, off, 32);
  __shared__ float red[8];
  const int lane = threadIdx.x & 31, wave = threadIdx.x >> 5;
  if (lane == 0) red[wave] = ss;
  __syncthreads();
  if (threadIdx.x == 0) {
    float s = 0.f;
    for (int i = 0; i < 8; ++i) s += red[i];
    red[0] = s;
  }
  __syncthreads();
  const float inv = rsqrtf(red[0] / (float)HIDN + 1e-6f);
  for (int i = threadIdx.x; i < HIDN; i += 256)
    O[(size_t)t * HIDN + i] = (bf16_t)(x[i] * inv * W[i]);
}

// ---------- generic bf16 WMMA GEMM: C[M,N] = A[M,K] * W[N,K]^T ----------
// Workgroup tile 64(M) x 128(N); 4 waves, each 32x64 (2x4 frags).
// Weight tiles are TDM-staged into LDS, double buffered (TENSORcnt).
// EPI 0: f32 store   2: f32 res+acc   3: atomicAdd(C, comb[row*NEXP+e]*acc)
template <int EPI>
__global__ __launch_bounds__(128)
void gemm_bf16_kernel(const bf16_t* __restrict__ Abase, const bf16_t* __restrict__ Wbase,
                      float* __restrict__ C, const float* __restrict__ Res,
                      const float* __restrict__ ScaleBase, int N, int K,
                      size_t aBatch, size_t wBatch) {
  __shared__ bf16_t Bs[2][128 * 64];
  const int lane = threadIdx.x & 31;
  const int wave = threadIdx.x >> 5;
  const int e = blockIdx.z;
  const bf16_t* A = Abase + (size_t)e * aBatch;
  const bf16_t* W = Wbase + (size_t)e * wBatch;
  const int m0 = blockIdx.y * 64 + (wave >> 1) * 32;
  const int nBlock = blockIdx.x * 128;
  const int wn = (wave & 1) * 64;  // local col base (64 wide per wave)

  v8f acc[2][4] = {};
  if (wave == 0)
    tdm_load_2d((unsigned)(size_t)&Bs[0][0], W + (size_t)nBlock * K, 128, 64, K, N);
  int cur = 0;
  for (int kc = 0; kc < K; kc += 64) {
    if (wave == 0) {
      if (kc + 64 < K) {
        tdm_load_2d((unsigned)(size_t)&Bs[cur ^ 1][0],
                    W + (size_t)nBlock * K + kc + 64, 128, 64, K, N);
        __builtin_amdgcn_s_wait_tensorcnt(1);
      } else {
        __builtin_amdgcn_s_wait_tensorcnt(0);
      }
    }
    __syncthreads();
    const bf16_t* Bc = &Bs[cur][0];
#pragma unroll
    for (int kk = 0; kk < 64; kk += 32) {
      v16bf a0 = load_a_frag(A, K, m0,      kc + kk, lane);
      v16bf a1 = load_a_frag(A, K, m0 + 16, kc + kk, lane);
#pragma unroll
      for (int j = 0; j < 4; ++j) {
        v16bf b = load_b_lds(Bc, wn + 16 * j, kk, lane);
        acc[0][j] = WMMA_BF16(a0, b, acc[0][j]);
        acc[1][j] = WMMA_BF16(a1, b, acc[1][j]);
      }
    }
    __syncthreads();
    cur ^= 1;
  }
  const int row15 = lane & 15, half = lane >> 4;
#pragma unroll
  for (int mi = 0; mi < 2; ++mi)
#pragma unroll
    for (int j = 0; j < 4; ++j)
#pragma unroll
      for (int r = 0; r < 8; ++r) {
        const int row = m0 + 16 * mi + r + 8 * half;
        const int col = nBlock + wn + 16 * j + row15;
        const size_t off = (size_t)row * N + col;
        const float v = acc[mi][j][r];
        if constexpr (EPI == 0) C[off] = v;
        else if constexpr (EPI == 2) C[off] = Res[off] + v;
        else atomicAdd(&C[off], ScaleBase[(size_t)row * NEXP + e] * v);
      }
}

// ---------------- RoPE on Q,K (f32 in, bf16 out) ----------------
__global__ __launch_bounds__(512)
void rope_kernel(const float* __restrict__ Qf, const float* __restrict__ Kf,
                 const int* __restrict__ pos, bf16_t* __restrict__ Qb,
                 bf16_t* __restrict__ Kb) {
  const int t = blockIdx.x;
  const int j = threadIdx.x;          // 0..511
  const int hh = j >> 5, d = j & 31;  // head, rot index
  const float p = (float)pos[t];
  const float inv = powf(1.0e6f, -(float)(2 * d) / (float)HDIM);
  const float fr = p * inv;
  const float c = cosf(fr), s = sinf(fr);
  {
    const float* q0 = Qf + (size_t)t * (NHEAD * HDIM) + hh * HDIM;
    bf16_t* o = Qb + (size_t)t * (NHEAD * HDIM) + hh * HDIM;
    const float x1 = q0[d], x2 = q0[d + 32];
    o[d]      = (bf16_t)(x1 * c - x2 * s);
    o[d + 32] = (bf16_t)(x2 * c + x1 * s);
  }
  if (j < NKVH * 32) {
    const float* k0 = Kf + (size_t)t * (NKVH * HDIM) + hh * HDIM;
    bf16_t* o = Kb + (size_t)t * (NKVH * HDIM) + hh * HDIM;
    const float x1 = k0[d], x2 = k0[d + 32];
    o[d]      = (bf16_t)(x1 * c - x2 * s);
    o[d + 32] = (bf16_t)(x2 * c + x1 * s);
  }
}

// ---------------- V transpose: [t][kv*64+d] f32 -> [kv*64+d][t] bf16 ----------------
__global__ void vtrans_kernel(const float* __restrict__ V, bf16_t* __restrict__ Vt) {
  const int idx = blockIdx.x * 256 + threadIdx.x;
  const int t = idx >> 8, c = idx & 255;
  Vt[(size_t)c * SEQ + t] = (bf16_t)V[(size_t)t * (NKVH * HDIM) + c];
}

// ---------------- causal flash attention (WMMA), 1 wave = 16 queries ----------------
__global__ __launch_bounds__(128)
void flash_attn_kernel(const bf16_t* __restrict__ Q, const bf16_t* __restrict__ Kb,
                       const bf16_t* __restrict__ Vt, bf16_t* __restrict__ O) {
  const int lane = threadIdx.x & 31;
  const int wv = threadIdx.x >> 5;
  const int h = blockIdx.y;
  const int kv = h >> 2;  // NHEAD/NKVH = 4
  const int t0 = (blockIdx.x * 4 + wv) * 16;
  __shared__ __bf16 ldsP[4][16 * 32];
  __bf16* myLds = ldsP[wv];
  const int row15 = lane & 15, half = lane >> 4;

  float m[8], l[8];
#pragma unroll
  for (int r = 0; r < 8; ++r) { m[r] = -1e30f; l[r] = 0.f; }
  v8f o[4] = {};

  for (int s0 = 0; s0 < t0 + 16; s0 += 32) {
    v8f sf[2] = {};
#pragma unroll
    for (int dk = 0; dk < HDIM; dk += 32) {
      v16bf aq = load_a_frag(Q, NHEAD * HDIM, t0, h * HDIM + dk, lane);
      v16bf b0 = load_b_frag(Kb, NKVH * HDIM, s0,      kv * HDIM + dk, lane);
      v16bf b1 = load_b_frag(Kb, NKVH * HDIM, s0 + 16, kv * HDIM + dk, lane);
      sf[0] = WMMA_BF16(aq, b0, sf[0]);
      sf[1] = WMMA_BF16(aq, b1, sf[1]);
    }
    float p[2][8];
#pragma unroll
    for (int f = 0; f < 2; ++f) {
      const int sidx = s0 + 16 * f + row15;
#pragma unroll
      for (int r = 0; r < 8; ++r) {
        const int tidx = t0 + r + 8 * half;
        const float v = sf[f][r] * 0.125f;  // 1/sqrt(64)
        p[f][r] = (sidx <= tidx) ? v : -1e30f;
      }
    }
    float alpha[8];
#pragma unroll
    for (int r = 0; r < 8; ++r) {
      float vmax = fmaxf(p[0][r], p[1][r]);
#pragma unroll
      for (int off = 1; off < 16; off <<= 1) vmax = fmaxf(vmax, __shfl_xor(vmax, off, 32));
      const float mn = fmaxf(m[r], vmax);
      alpha[r] = __expf(m[r] - mn);
      m[r] = mn;
      const float e0 = __expf(p[0][r] - mn), e1 = __expf(p[1][r] - mn);
      p[0][r] = e0; p[1][r] = e1;
      float rs = e0 + e1;
#pragma unroll
      for (int off = 1; off < 16; off <<= 1) rs += __shfl_xor(rs, off, 32);
      l[r] = l[r] * alpha[r] + rs;
    }
#pragma unroll
    for (int j = 0; j < 4; ++j)
#pragma unroll
      for (int r = 0; r < 8; ++r) o[j][r] *= alpha[r];
    // transpose P (C-layout -> A-layout) via per-wave LDS bounce (same-wave
    // LDS ops are in-order, no barrier needed)
#pragma unroll
    for (int f = 0; f < 2; ++f)
#pragma unroll
      for (int r = 0; r < 8; ++r)
        myLds[(r + 8 * half) * 32 + 16 * f + row15] = (__bf16)p[f][r];
    v16bf pa;
    {
      const __bf16* src = &myLds[row15 * 32 + 16 * half];
#pragma unroll
      for (int i = 0; i < 8; ++i) { pa[i] = src[i]; pa[8 + i] = src[8 + i]; }
    }
#pragma unroll
    for (int j = 0; j < 4; ++j) {
      v16bf bv = load_b_frag(Vt + (size_t)kv * HDIM * SEQ, SEQ, 16 * j, s0, lane);
      o[j] = WMMA_BF16(pa, bv, o[j]);
    }
  }
#pragma unroll
  for (int j = 0; j < 4; ++j)
#pragma unroll
    for (int r = 0; r < 8; ++r) {
      const int trow = t0 + r + 8 * half;
      O[(size_t)trow * (NHEAD * HDIM) + h * HDIM + 16 * j + row15] =
          (bf16_t)(o[j][r] / l[r]);
    }
}

// ---------------- gating: softmax over 8 logits, top-2, renormalize ----------------
__global__ __launch_bounds__(32)
void gate_kernel(const bf16_t* __restrict__ X, const float* __restrict__ GW,
                 float* __restrict__ comb) {
  const int t = blockIdx.x, lane = threadIdx.x;
  float acc[NEXP] = {};
  for (int k = lane; k < HIDN; k += 32) {
    const float xv = (float)X[(size_t)t * HIDN + k];
#pragma unroll
    for (int e = 0; e < NEXP; ++e) acc[e] += xv * GW[(size_t)e * HIDN + k];
  }
#pragma unroll
  for (int e = 0; e < NEXP; ++e)
#pragma unroll
    for (int off = 16; off; off >>= 1) acc[e] += __shfl_xor(acc[e], off, 32);
  if (lane == 0) {
    float mx = acc[0];
    for (int e = 1; e < NEXP; ++e) mx = fmaxf(mx, acc[e]);
    float pr[NEXP], den = 0.f;
    for (int e = 0; e < NEXP; ++e) { pr[e] = __expf(acc[e] - mx); den += pr[e]; }
    for (int e = 0; e < NEXP; ++e) pr[e] /= den;
    int i0 = 0;
    for (int e = 1; e < NEXP; ++e) if (pr[e] > pr[i0]) i0 = e;
    int i1 = (i0 == 0) ? 1 : 0;
    for (int e = 0; e < NEXP; ++e) if (e != i0 && pr[e] > pr[i1]) i1 = e;
    const float s = pr[i0] + pr[i1];
    for (int e = 0; e < NEXP; ++e)
      comb[(size_t)t * NEXP + e] = (e == i0) ? pr[i0] / s : ((e == i1) ? pr[i1] / s : 0.f);
  }
}

// ---- MoE gate/up: hh = silu(x w1^T) * (x w3^T); TDM-staged w1/w3 tiles ----
__global__ __launch_bounds__(128)
void moe_gateup_kernel(const bf16_t* __restrict__ X, const bf16_t* __restrict__ W1b,
                       const bf16_t* __restrict__ W3b, bf16_t* __restrict__ HH) {
  __shared__ bf16_t B1s[2][128 * 64];
  __shared__ bf16_t B3s[2][128 * 64];
  const int e = blockIdx.z;
  const bf16_t* w1 = W1b + (size_t)e * INTERN * HIDN;
  const bf16_t* w3 = W3b + (size_t)e * INTERN * HIDN;
  bf16_t* hh = HH + (size_t)e * SEQ * INTERN;
  const int lane = threadIdx.x & 31;
  const int wave = threadIdx.x >> 5;
  const int m0 = blockIdx.y * 64 + (wave >> 1) * 32;
  const int nBlock = blockIdx.x * 128;
  const int wn = (wave & 1) * 64;

  v8f c1[2][4] = {}, c3[2][4] = {};
  if (wave == 0) {
    tdm_load_2d((unsigned)(size_t)&B1s[0][0], w1 + (size_t)nBlock * HIDN, 128, 64, HIDN, INTERN);
    tdm_load_2d((unsigned)(size_t)&B3s[0][0], w3 + (size_t)nBlock * HIDN, 128, 64, HIDN, INTERN);
  }
  int cur = 0;
  for (int kc = 0; kc < HIDN; kc += 64) {
    if (wave == 0) {
      if (kc + 64 < HIDN) {
        tdm_load_2d((unsigned)(size_t)&B1s[cur ^ 1][0],
                    w1 + (size_t)nBlock * HIDN + kc + 64, 128, 64, HIDN, INTERN);
        tdm_load_2d((unsigned)(size_t)&B3s[cur ^ 1][0],
                    w3 + (size_t)nBlock * HIDN + kc + 64, 128, 64, HIDN, INTERN);
        __builtin_amdgcn_s_wait_tensorcnt(2);
      } else {
        __builtin_amdgcn_s_wait_tensorcnt(0);
      }
    }
    __syncthreads();
    const bf16_t* B1c = &B1s[cur][0];
    const bf16_t* B3c = &B3s[cur][0];
#pragma unroll
    for (int kk = 0; kk < 64; kk += 32) {
      v16bf a0 = load_a_frag(X, HIDN, m0,      kc + kk, lane);
      v16bf a1 = load_a_frag(X, HIDN, m0 + 16, kc + kk, lane);
#pragma unroll
      for (int j = 0; j < 4; ++j) {
        v16bf bu = load_b_lds(B1c, wn + 16 * j, kk, lane);
        c1[0][j] = WMMA_BF16(a0, bu, c1[0][j]);
        c1[1][j] = WMMA_BF16(a1, bu, c1[1][j]);
        v16bf bg = load_b_lds(B3c, wn + 16 * j, kk, lane);
        c3[0][j] = WMMA_BF16(a0, bg, c3[0][j]);
        c3[1][j] = WMMA_BF16(a1, bg, c3[1][j]);
      }
    }
    __syncthreads();
    cur ^= 1;
  }
  const int row15 = lane & 15, half = lane >> 4;
#pragma unroll
  for (int mi = 0; mi < 2; ++mi)
#pragma unroll
    for (int j = 0; j < 4; ++j)
#pragma unroll
      for (int r = 0; r < 8; ++r) {
        const int row = m0 + 16 * mi + r + 8 * half;
        const int col = nBlock + wn + 16 * j + row15;
        const float h1 = c1[mi][j][r];
        const float h3 = c3[mi][j][r];
        const float silu = h1 / (1.f + __expf(-h1));
        hh[(size_t)row * INTERN + col] = (bf16_t)(silu * h3);
      }
}

// =============================== host launcher ===============================
extern "C" void kernel_launch(void* const* d_in, const int* in_sizes, int n_in,
                              void* d_out, int out_size, void* d_ws, size_t ws_size,
                              hipStream_t stream) {
  (void)in_sizes; (void)n_in; (void)out_size; (void)ws_size;
  const float* h      = (const float*)d_in[0];
  const int*   pos    = (const int*)d_in[1];
  const float* wq     = (const float*)d_in[2];
  const float* wk     = (const float*)d_in[3];
  const float* wv     = (const float*)d_in[4];
  const float* wo     = (const float*)d_in[5];
  const float* gate_w = (const float*)d_in[6];
  const float* w1     = (const float*)d_in[7];
  const float* w2     = (const float*)d_in[8];
  const float* w3     = (const float*)d_in[9];
  const float* ln1_w  = (const float*)d_in[10];
  const float* ln2_w  = (const float*)d_in[11];
  float* out = (float*)d_out;

  // ---- carve workspace ----
  char* p = (char*)d_ws;
  auto take = [&](size_t bytes) -> char* {
    char* r = p;
    p += (bytes + 255) & ~(size_t)255;
    return r;
  };
  bf16_t* wq_b  = (bf16_t*)take((size_t)NHEAD * HDIM * HIDN * 2);
  bf16_t* wk_b  = (bf16_t*)take((size_t)NKVH * HDIM * HIDN * 2);
  bf16_t* wv_b  = (bf16_t*)take((size_t)NKVH * HDIM * HIDN * 2);
  bf16_t* wo_b  = (bf16_t*)take((size_t)HIDN * NHEAD * HDIM * 2);
  bf16_t* w1_b  = (bf16_t*)take((size_t)NEXP * INTERN * HIDN * 2);
  bf16_t* w2_b  = (bf16_t*)take((size_t)NEXP * HIDN * INTERN * 2);
  bf16_t* w3_b  = (bf16_t*)take((size_t)NEXP * INTERN * HIDN * 2);
  bf16_t* x1_b  = (bf16_t*)take((size_t)SEQ * HIDN * 2);
  bf16_t* x2_b  = (bf16_t*)take((size_t)SEQ * HIDN * 2);
  float*  q_f   = (float*)take((size_t)SEQ * NHEAD * HDIM * 4);
  float*  k_f   = (float*)take((size_t)SEQ * NKVH * HDIM * 4);
  float*  v_f   = (float*)take((size_t)SEQ * NKVH * HDIM * 4);
  bf16_t* q_b   = (bf16_t*)take((size_t)SEQ * NHEAD * HDIM * 2);
  bf16_t* k_b   = (bf16_t*)take((size_t)SEQ * NKVH * HDIM * 2);
  bf16_t* vt_b  = (bf16_t*)take((size_t)NKVH * HDIM * SEQ * 2);
  bf16_t* at_b  = (bf16_t*)take((size_t)SEQ * NHEAD * HDIM * 2);
  float*  comb  = (float*)take((size_t)SEQ * NEXP * 4);
  bf16_t* hh_b  = (bf16_t*)take((size_t)NEXP * SEQ * INTERN * 2);

  // ---- weight conversions (once; ~0.5 GB traffic ≈ 23 us at 23.3 TB/s) ----
  f32_to_bf16_kernel<<<4096, 256, 0, stream>>>(wq, wq_b, (size_t)NHEAD * HDIM * HIDN);
  f32_to_bf16_kernel<<<1024, 256, 0, stream>>>(wk, wk_b, (size_t)NKVH * HDIM * HIDN);
  f32_to_bf16_kernel<<<1024, 256, 0, stream>>>(wv, wv_b, (size_t)NKVH * HDIM * HIDN);
  f32_to_bf16_kernel<<<4096, 256, 0, stream>>>(wo, wo_b, (size_t)HIDN * NHEAD * HDIM);
  f32_to_bf16_kernel<<<8192, 256, 0, stream>>>(w1, w1_b, (size_t)NEXP * INTERN * HIDN);
  f32_to_bf16_kernel<<<8192, 256, 0, stream>>>(w2, w2_b, (size_t)NEXP * HIDN * INTERN);
  f32_to_bf16_kernel<<<8192, 256, 0, stream>>>(w3, w3_b, (size_t)NEXP * INTERN * HIDN);

  // ---- attention block ----
  rmsnorm_kernel<<<SEQ, 256, 0, stream>>>(h, ln1_w, x1_b);
  gemm_bf16_kernel<0><<<dim3((NHEAD * HDIM) / 128, SEQ / 64), 128, 0, stream>>>(
      x1_b, wq_b, q_f, nullptr, nullptr, NHEAD * HDIM, HIDN, 0, 0);
  gemm_bf16_kernel<0><<<dim3((NKVH * HDIM) / 128, SEQ / 64), 128, 0, stream>>>(
      x1_b, wk_b, k_f, nullptr, nullptr, NKVH * HDIM, HIDN, 0, 0);
  gemm_bf16_kernel<0><<<dim3((NKVH * HDIM) / 128, SEQ / 64), 128, 0, stream>>>(
      x1_b, wv_b, v_f, nullptr, nullptr, NKVH * HDIM, HIDN, 0, 0);
  rope_kernel<<<SEQ, 512, 0, stream>>>(q_f, k_f, pos, q_b, k_b);
  vtrans_kernel<<<SEQ, 256, 0, stream>>>(v_f, vt_b);
  flash_attn_kernel<<<dim3(SEQ / 64, NHEAD), 128, 0, stream>>>(q_b, k_b, vt_b, at_b);
  // out = h + attn @ wo^T   (out doubles as h_mid residual)
  gemm_bf16_kernel<2><<<dim3(HIDN / 128, SEQ / 64), 128, 0, stream>>>(
      at_b, wo_b, out, h, nullptr, HIDN, NHEAD * HDIM, 0, 0);

  // ---- MoE block ----
  rmsnorm_kernel<<<SEQ, 256, 0, stream>>>(out, ln2_w, x2_b);
  gate_kernel<<<SEQ, 32, 0, stream>>>(x2_b, gate_w, comb);
  moe_gateup_kernel<<<dim3(INTERN / 128, SEQ / 64, NEXP), 128, 0, stream>>>(
      x2_b, w1_b, w3_b, hh_b);
  // out += comb[t,e] * (hh_e @ w2_e^T)
  gemm_bf16_kernel<3><<<dim3(HIDN / 128, SEQ / 64, NEXP), 128, 0, stream>>>(
      hh_b, w2_b, out, nullptr, comb, HIDN, INTERN,
      (size_t)SEQ * INTERN, (size_t)HIDN * INTERN);
}
